// WhiteningLayer_68616397521494
// MI455X (gfx1250) — compile-verified, matrix operand order checked by reading, not run on hardware
//
#include <hip/hip_runtime.h>
#include <hip/hip_bf16.h>

typedef __attribute__((ext_vector_type(16))) __bf16 v16bf;
typedef __attribute__((ext_vector_type(8)))  __bf16 v8bf;
typedef __attribute__((ext_vector_type(8)))  float  v8f;

#define L_DIM   1024
#define B_DIM   8
#define H_DIM   2048
#define DK      128
#define NPAIR   128          // B*HEADS
#define BH      (B_DIM * H_DIM)

__device__ __forceinline__ unsigned short f2bf(float f) {
    unsigned int u = __float_as_uint(f);
    unsigned int r = u + 0x7FFFu + ((u >> 16) & 1u);   // round-to-nearest-even
    return (unsigned short)(r >> 16);
}

// ---------------------------------------------------------------------------
// Kernel 1: qrow[p][j] = q_bh . Wq[:,j] + bias[j]   (fp32, tiny)
// ---------------------------------------------------------------------------
__global__ __launch_bounds__(256) void qrow_kernel(
    const float* __restrict__ query, const float* __restrict__ attn_w,
    const float* __restrict__ bias,  float* __restrict__ qrow)
{
    __shared__ float qs[DK];
    const int p = blockIdx.x;            // pair
    const int b = p >> 4, h = p & 15;
    const int t = threadIdx.x;
    if (t < DK) qs[t] = query[b * H_DIM + h * DK + t];
    __syncthreads();
    for (int jj = 0; jj < 8; ++jj) {
        const int j = t + jj * 256;
        float acc = bias[j];
        #pragma unroll 4
        for (int i = 0; i < DK; ++i)
            acc += qs[i] * attn_w[i * H_DIM + j];    // Wq = attn_w[0:128]
        qrow[p * H_DIM + j] = acc;
    }
}

// ---------------------------------------------------------------------------
// Kernel 2: pre-pack Wk (attn_w rows 128..255) into bf16 B-operand fragments.
// Fragment (jt, kb): 32 lanes x 16 bf16.  Lane<16: N=lane, K = kb*32 + (e<8? e : e+8)
//                                         Lane>=16: N=lane-16, K = kb*32+8 + (e<8? e : e+8)
// ---------------------------------------------------------------------------
__global__ __launch_bounds__(128) void wkpack_kernel(
    const float* __restrict__ attn_w, unsigned short* __restrict__ wkp)
{
    const int jt   = blockIdx.x;          // 0..127 (16-column j tile)
    const int t    = threadIdx.x;         // 0..127
    const int kb   = t >> 5;              // 0..3
    const int lane = t & 31;
    const int n    = lane & 15;
    const int koff = (lane < 16) ? 0 : 8;
    unsigned short* dst = wkp + ((size_t)(jt * 4 + kb) * 32 + lane) * 16;
    #pragma unroll
    for (int e = 0; e < 16; ++e) {
        const int K = kb * 32 + koff + ((e < 8) ? e : (e + 8));
        const float w = attn_w[(size_t)(DK + K) * H_DIM + jt * 16 + n];
        dst[e] = f2bf(w);
    }
}

// ---------------------------------------------------------------------------
// Kernel 3: fused K@Wk GEMM + relu(qrow + .) * score reduction -> logits
// grid = 128 pairs * 8 L-chunks, block = 256 (8 waves), wave owns 16 L rows.
// ---------------------------------------------------------------------------
__global__ __launch_bounds__(256) void gemm_logits_kernel(
    const float* __restrict__ feed,  const unsigned short* __restrict__ wkp,
    const float* __restrict__ qrow,  const float* __restrict__ score,
    float* __restrict__ logits)
{
    __shared__ unsigned short Kbf[128][136];      // 128x128 bf16 K-chunk, +8 pad (bank-conflict free)

    const int blk   = blockIdx.x;
    const int pair  = blk >> 3;
    const int lc    = blk & 7;
    const int b     = pair >> 4, h = pair & 15;
    const int lbase = lc * 128;
    const int t     = threadIdx.x;

    // ---- stage K chunk (fp32 -> bf16) into LDS: 16384 floats, float4 x 16/thread
    const float* fbase = feed + (size_t)lbase * BH + b * H_DIM + h * DK;
    for (int it = 0; it < 16; ++it) {
        const int idx  = t + it * 256;         // float4 index 0..4095
        const int row  = idx >> 5;             // 32 float4 per 128-float row
        const int col4 = (idx & 31) * 4;
        const float4 f = *(const float4*)(fbase + (size_t)row * BH + col4);
        Kbf[row][col4 + 0] = f2bf(f.x);
        Kbf[row][col4 + 1] = f2bf(f.y);
        Kbf[row][col4 + 2] = f2bf(f.z);
        Kbf[row][col4 + 3] = f2bf(f.w);
    }
    __syncthreads();

    const int wave = t >> 5;
    const int lane = t & 31;
    const int M    = lane & 15;
    const int row  = wave * 16 + M;
    const int koff = (lane < 16) ? 0 : 8;

    // ---- build 4 A fragments once, keep in VGPRs for the whole j sweep
    v16bf a[4];
    #pragma unroll
    for (int kb = 0; kb < 4; ++kb) {
        const v8bf lo = *reinterpret_cast<const v8bf*>(&Kbf[row][kb * 32 + koff]);
        const v8bf hi = *reinterpret_cast<const v8bf*>(&Kbf[row][kb * 32 + koff + 16]);
        a[kb] = __builtin_shufflevector(lo, hi, 0,1,2,3,4,5,6,7,8,9,10,11,12,13,14,15);
    }

    float logacc[8] = {0.f,0.f,0.f,0.f,0.f,0.f,0.f,0.f};
    const v16bf* wkpv = reinterpret_cast<const v16bf*>(wkp);
    const float* qr   = qrow + pair * H_DIM;

    for (int jt = 0; jt < 128; ++jt) {
        v8f c = {0.f,0.f,0.f,0.f,0.f,0.f,0.f,0.f};
        #pragma unroll
        for (int kb = 0; kb < 4; ++kb) {
            const v16bf bfrag = wkpv[(jt * 4 + kb) * 32 + lane];
            c = __builtin_amdgcn_wmma_f32_16x16x32_bf16(
                    false, a[kb], false, bfrag, (short)0, c, false, false);
        }
        const int j   = jt * 16 + (lane & 15);
        const float q = qr[j];
        const float s = score[j];
        #pragma unroll
        for (int r = 0; r < 8; ++r) {
            float e = c[r] + q;
            e = e > 0.f ? e : 0.f;
            logacc[r] += e * s;                 // fused energy @ score
        }
    }

    // ---- reduce over the 16 N-columns held by each lane half
    #pragma unroll
    for (int r = 0; r < 8; ++r) {
        float v = logacc[r];
        v += __shfl_xor(v, 1, 32);
        v += __shfl_xor(v, 2, 32);
        v += __shfl_xor(v, 4, 32);
        v += __shfl_xor(v, 8, 32);
        logacc[r] = v;
    }
    // C layout: lanes 0-15 -> M=r, lanes 16-31 -> M=r+8
    if ((lane & 15) == 0) {
        const int mofs = (lane < 16) ? 0 : 8;
        #pragma unroll
        for (int r = 0; r < 8; ++r)
            logits[pair * L_DIM + lbase + wave * 16 + mofs + r] = logacc[r];
    }
}

// ---------------------------------------------------------------------------
// Kernel 4: softmax over L + a @ V  (streams values at HBM bandwidth)
// ---------------------------------------------------------------------------
__global__ __launch_bounds__(256) void softmax_av_kernel(
    const float* __restrict__ values, const float* __restrict__ logits,
    float* __restrict__ out)
{
    __shared__ float al[L_DIM];
    __shared__ float red[256];
    __shared__ float part[2][DK];

    const int pair = blockIdx.x;
    const int b = pair >> 4, h = pair & 15;
    const int t = threadIdx.x;
    const float* lg = logits + pair * L_DIM;

    const float l0 = lg[t], l1 = lg[t + 256], l2 = lg[t + 512], l3 = lg[t + 768];
    red[t] = fmaxf(fmaxf(l0, l1), fmaxf(l2, l3));
    __syncthreads();
    for (int s = 128; s > 0; s >>= 1) {
        if (t < s) red[t] = fmaxf(red[t], red[t + s]);
        __syncthreads();
    }
    const float mx = red[0];
    __syncthreads();

    const float e0 = __expf(l0 - mx), e1 = __expf(l1 - mx);
    const float e2 = __expf(l2 - mx), e3 = __expf(l3 - mx);
    al[t] = e0; al[t + 256] = e1; al[t + 512] = e2; al[t + 768] = e3;
    red[t] = e0 + e1 + e2 + e3;
    __syncthreads();
    for (int s = 128; s > 0; s >>= 1) {
        if (t < s) red[t] += red[t + s];
        __syncthreads();
    }
    const float inv = 1.0f / red[0];

    const int i = t & 127, g = t >> 7;
    const float* vbase = values + b * H_DIM + h * DK + i;
    float acc = 0.f;
    #pragma unroll 4
    for (int l = g; l < L_DIM; l += 2)
        acc += al[l] * vbase[(size_t)l * BH];
    part[g][i] = acc;
    __syncthreads();
    if (t < DK)
        out[b * H_DIM + h * DK + t] = (part[0][t] + part[1][t]) * inv;
}

// ---------------------------------------------------------------------------
extern "C" void kernel_launch(void* const* d_in, const int* in_sizes, int n_in,
                              void* d_out, int out_size, void* d_ws, size_t ws_size,
                              hipStream_t stream)
{
    const float* query  = (const float*)d_in[0];   // (8, 2048)
    const float* feed   = (const float*)d_in[1];   // (1024, 8, 2048)
    const float* values = (const float*)d_in[2];   // (1024, 8, 2048)
    const float* attn_w = (const float*)d_in[3];   // (256, 2048)
    const float* bias   = (const float*)d_in[4];   // (2048,)
    const float* score  = (const float*)d_in[5];   // (2048, 1)
    float* out = (float*)d_out;                    // (1, 8, 2048)

    // workspace layout: qrow (1MB) | logits (512KB) | WkPack bf16 (512KB)
    float* qrow   = (float*)d_ws;                        // 128*2048 f32
    float* logits = qrow + NPAIR * H_DIM;                // 128*1024 f32
    unsigned short* wkp = (unsigned short*)(logits + NPAIR * L_DIM); // 128*4*32*16 bf16

    qrow_kernel<<<NPAIR, 256, 0, stream>>>(query, attn_w, bias, qrow);
    wkpack_kernel<<<128, 128, 0, stream>>>(attn_w, wkp);
    gemm_logits_kernel<<<NPAIR * 8, 256, 0, stream>>>(feed, wkp, qrow, score, logits);
    softmax_av_kernel<<<NPAIR, 256, 0, stream>>>(values, logits, out);
}
